// GATConv_42649025249476
// MI455X (gfx1250) — compile-verified, matrix operand order checked by reading, not run on hardware
//
#include <hip/hip_runtime.h>
#include <hip/hip_bf16.h>

#define NN 8192
#define NEMB 128
#define NHID 64
#define NEG_BIG -9000000000000000.0f
#define SLOPE 0.01f

typedef __attribute__((ext_vector_type(16))) __bf16 v16bf;
typedef __attribute__((ext_vector_type(8)))  __bf16 v8bf;
typedef __attribute__((ext_vector_type(8)))  float  v8f;
typedef __attribute__((ext_vector_type(4)))  unsigned v4u;
typedef __attribute__((ext_vector_type(8)))  int      v8i;
typedef __attribute__((ext_vector_type(4)))  int      v4i;
typedef __attribute__((address_space(3)))    __bf16   lds_bf16;

// ---------------- K1: h = x @ W^T + bias  (fp32, tiny: 0.13 GFLOP) -------------
__global__ __launch_bounds__(256) void k_hidden(const float* __restrict__ x,
                                                const float* __restrict__ W,
                                                const float* __restrict__ bias,
                                                float* __restrict__ h) {
  __shared__ float sW[NHID][NEMB];   // 32 KB
  __shared__ float sx[4][NEMB];      // 2 KB
  int tid = threadIdx.x;
  for (int i = tid; i < NHID * NEMB; i += 256) sW[i / NEMB][i % NEMB] = W[i];
  int r0 = blockIdx.x * 4;
  for (int i = tid; i < 4 * NEMB; i += 256)
    sx[i / NEMB][i % NEMB] = x[(size_t)(r0 + i / NEMB) * NEMB + (i % NEMB)];
  __syncthreads();
  int lr = tid >> 6;
  int f  = tid & 63;
  float acc = bias[f];
  #pragma unroll 8
  for (int k = 0; k < NEMB; ++k) acc += sx[lr][k] * sW[f][k];
  h[(size_t)(r0 + lr) * NHID + f] = acc;
}

// ---- K2: s_src/s_dst + emit bf16 transposed copy hT[f][r] for the WMMA B tiles
__global__ __launch_bounds__(256) void k_scores(const float* __restrict__ h,
                                                const float* __restrict__ aW,
                                                const float* __restrict__ aB,
                                                float* __restrict__ s_src,
                                                float* __restrict__ s_dst,
                                                __bf16* __restrict__ hT) {
  __shared__ float sa[2 * NHID];
  int tid = threadIdx.x;
  if (tid < 2 * NHID) sa[tid] = aW[tid];
  __syncthreads();
  int r = blockIdx.x * 256 + tid;
  const float* hr = h + (size_t)r * NHID;
  float ss = 0.f, sd = 0.f;
  #pragma unroll 8
  for (int f = 0; f < NHID; ++f) {
    float v = hr[f];
    ss += v * sa[f];
    sd += v * sa[NHID + f];
    hT[(size_t)f * NN + r] = (__bf16)v;   // coalesced across threads per f
  }
  s_src[r] = ss + aB[0];
  s_dst[r] = sd;
}

// ---------------- K3: per-row nonzero counts of adj ----------------------------
__global__ __launch_bounds__(256) void k_rowcount(const int* __restrict__ adj,
                                                  unsigned* __restrict__ rowCnt) {
  int wave = threadIdx.x >> 5, lane = threadIdx.x & 31;
  int r = blockIdx.x * 8 + wave;
  const int4* arow = (const int4*)(adj + (size_t)r * NN);
  int cnt = 0;
  for (int i = lane; i < NN / 4; i += 32) {
    int4 a = arow[i];
    cnt += (a.x > 0) + (a.y > 0) + (a.z > 0) + (a.w > 0);
  }
  #pragma unroll
  for (int off = 16; off; off >>= 1) cnt += __shfl_xor(cnt, off, 32);
  if (lane == 0) rowCnt[r] = (unsigned)cnt;
}

// ---------------- K4: exclusive scan of row counts (single wave) ---------------
__global__ __launch_bounds__(32) void k_scan(const unsigned* __restrict__ rowCnt,
                                             unsigned* __restrict__ rowBase) {
  int lane = threadIdx.x;
  unsigned running = 0;
  for (int c = 0; c < NN / 32; ++c) {
    unsigned v = rowCnt[c * 32 + lane];
    unsigned incl = v;
    #pragma unroll
    for (int off = 1; off < 32; off <<= 1) {
      unsigned t = __shfl_up(incl, off, 32);
      if (lane >= off) incl += t;
    }
    rowBase[c * 32 + lane] = running + incl - v;
    running += __shfl(incl, 31, 32);
  }
}

// raw edge score for global edge-rank m (always safe: m < 2^26 -> indices < 8192)
__device__ __forceinline__ float edge_e(unsigned m, const float* __restrict__ s_src,
                                        const float* __restrict__ s_dst) {
  float e = s_src[m >> 13] + s_dst[m & (NN - 1)];
  return (e > 0.f) ? e : e * SLOPE;
}

// ---------------- K5: fused rank/softmax + P·h via bf16 WMMA -------------------
// block = 512 threads = 16 wave32 = 16 rows. Super-chunk = 256 columns.
#define SUP 256
#define STR 264   // 256 data + 8 pad elements (TDM pad: 128 DWORDs + 4 DWORDs)
__global__ __launch_bounds__(512) void k_gat_main(const int* __restrict__ adj,
                                                  const __bf16* __restrict__ hT,
                                                  const float* __restrict__ s_src,
                                                  const float* __restrict__ s_dst,
                                                  const unsigned* __restrict__ rowBase,
                                                  float* __restrict__ out) {
  __shared__ __align__(16) __bf16 Ht[NHID][STR]; // 33,792 B : h tile [feat][col] (TDM fills)
  __shared__ __align__(16) __bf16 Pt[16][STR];   //  8,448 B : P tile  [row][col]
  __shared__ float Red[16][8][32];               // 16,384 B : cross-wave C reduction

  const int tid  = threadIdx.x;
  const int wave = tid >> 5;
  const int lane = tid & 31;
  const int r    = blockIdx.x * 16 + wave;
  const int* arow = adj + (size_t)r * NN;
  const unsigned base = rowBase[r];
  const unsigned lt   = (1u << lane) - 1u;

  // ---- Phase A: per-row online max/sum over e (branch-free, int4 loads) ----
  float M = NEG_BIG, L = 0.f;
  unsigned cnt = 0;
  for (int c0 = 0; c0 < NN; c0 += 128) {
    int4 a4 = *(const int4*)(arow + c0 + lane * 4);
    int s0 = a4.x > 0, s1 = a4.y > 0, s2 = a4.z > 0, s3 = a4.w > 0;
    unsigned b0 = (unsigned)__ballot(s0), b1 = (unsigned)__ballot(s1);
    unsigned b2 = (unsigned)__ballot(s2), b3 = (unsigned)__ballot(s3);
    unsigned m0 = base + cnt +
                  __popc(b0 & lt) + __popc(b1 & lt) + __popc(b2 & lt) + __popc(b3 & lt);
    unsigned m1 = m0 + s0, m2 = m1 + s1, m3 = m2 + s2;
    // unconditional gathers (issue as one clause), then select
    float r0 = edge_e(m0, s_src, s_dst);
    float r1 = edge_e(m1, s_src, s_dst);
    float r2 = edge_e(m2, s_src, s_dst);
    float r3 = edge_e(m3, s_src, s_dst);
    float e0 = s0 ? r0 : NEG_BIG;
    float e1 = s1 ? r1 : NEG_BIG;
    float e2 = s2 ? r2 : NEG_BIG;
    float e3 = s3 ? r3 : NEG_BIG;
    cnt += __popc(b0) + __popc(b1) + __popc(b2) + __popc(b3);
    float cm = fmaxf(fmaxf(e0, e1), fmaxf(e2, e3));
    #pragma unroll
    for (int off = 16; off; off >>= 1) cm = fmaxf(cm, __shfl_xor(cm, off, 32));
    float nm = fmaxf(M, cm);
    float p = __expf(e0 - nm) + __expf(e1 - nm) + __expf(e2 - nm) + __expf(e3 - nm);
    #pragma unroll
    for (int off = 16; off; off >>= 1) p += __shfl_xor(p, off, 32);
    L = L * __expf(M - nm) + p;
    M = nm;
  }
  const float invL = 1.f / L;

  // ---- TDM descriptor pieces (uniform per block) ----
  const unsigned ldsHt = (unsigned)(unsigned long long)(lds_bf16*)&Ht[0][0];
  // group1: data_size=2B, pad_enable, pad_interval=128 DW (code 6), pad_amount=4 DW (code 3)
  const unsigned g1w0 = (1u << 16) | (1u << 20) | (6u << 22) | (3u << 25);

  // ---- Phase B: probabilities -> LDS bf16 tile, WMMA against TDM-staged h tile ----
  v8f acc = {0.f, 0.f, 0.f, 0.f, 0.f, 0.f, 0.f, 0.f};
  const int ft = wave & 3;   // feature tile 0..3
  const int kg = wave >> 2;  // k-group 0..3
  unsigned cntB = 0;
  for (int c0 = 0; c0 < NN; c0 += SUP) {
    // issue async tensor load of hT[0:64, c0:c0+256] into Ht (one wave, overlaps Pt fill)
    if (wave == 0) {
      unsigned long long ga = (unsigned long long)(const void*)(hT + c0);
      v4u g0 = { 1u,                                      // count=1 (valid user D#)
                 ldsHt,                                   // lds_addr
                 (unsigned)(ga & 0xffffffffu),            // global_addr[31:0]
                 (unsigned)((ga >> 32) & 0x1ffffffu) | 0x80000000u }; // addr[56:32]|type=2
      v8i g1 = { (int)g1w0,
                 (int)((NN & 0xffffu) << 16),             // tensor_dim0=8192 lo16
                 (int)((NN >> 16) | (NHID << 16)),        // dim0 hi16 | tensor_dim1=64
                 (int)(SUP << 16),                        // tile_dim0=256
                 (int)NHID,                               // tile_dim1=64, tile_dim2=0
                 (int)NN,                                 // tensor_dim0_stride=8192
                 0, 0 };
      v4i gz4 = { 0, 0, 0, 0 };
      v8i gz8 = { 0, 0, 0, 0, 0, 0, 0, 0 };
      __builtin_amdgcn_tensor_load_to_lds(g0, g1, gz4, gz4, gz8, 0);
    }
    // fill probability tile: wave fills its own row (two 128-col vector steps)
    #pragma unroll
    for (int j = 0; j < SUP / 128; ++j) {
      int4 a4 = *(const int4*)(arow + c0 + j * 128 + lane * 4);
      int s0 = a4.x > 0, s1 = a4.y > 0, s2 = a4.z > 0, s3 = a4.w > 0;
      unsigned b0 = (unsigned)__ballot(s0), b1 = (unsigned)__ballot(s1);
      unsigned b2 = (unsigned)__ballot(s2), b3 = (unsigned)__ballot(s3);
      unsigned m0 = base + cntB +
                    __popc(b0 & lt) + __popc(b1 & lt) + __popc(b2 & lt) + __popc(b3 & lt);
      unsigned m1 = m0 + s0, m2 = m1 + s1, m3 = m2 + s2;
      float r0 = edge_e(m0, s_src, s_dst);
      float r1 = edge_e(m1, s_src, s_dst);
      float r2 = edge_e(m2, s_src, s_dst);
      float r3 = edge_e(m3, s_src, s_dst);
      float e0 = s0 ? r0 : NEG_BIG;
      float e1 = s1 ? r1 : NEG_BIG;
      float e2 = s2 ? r2 : NEG_BIG;
      float e3 = s3 ? r3 : NEG_BIG;
      cntB += __popc(b0) + __popc(b1) + __popc(b2) + __popc(b3);
      union { __bf16 b[4]; uint2 u; } pk;
      pk.b[0] = (__bf16)(__expf(e0 - M) * invL);
      pk.b[1] = (__bf16)(__expf(e1 - M) * invL);
      pk.b[2] = (__bf16)(__expf(e2 - M) * invL);
      pk.b[3] = (__bf16)(__expf(e3 - M) * invL);
      *(uint2*)&Pt[wave][j * 128 + lane * 4] = pk.u;   // one ds_store_b64
    }
    if (wave == 0) __builtin_amdgcn_s_wait_tensorcnt(0);
    __syncthreads();

    // WMMA: C[16x16] += P[16x32] * h[32x16], fp32 accumulate
    const int row  = lane & 15;
    const int half = lane >> 4;
    #pragma unroll
    for (int s = 0; s < 2; ++s) {
      const int kBase = (kg * 2 + s) * 32;
      union { v16bf v; v8bf p[2]; } A, B;
      const __bf16* pa = &Pt[row][kBase + half * 8];
      A.p[0] = *(const v8bf*)pa;
      A.p[1] = *(const v8bf*)(pa + 16);
      const __bf16* pb = &Ht[ft * 16 + row][kBase + half * 8];
      B.p[0] = *(const v8bf*)pb;
      B.p[1] = *(const v8bf*)(pb + 16);
      acc = __builtin_amdgcn_wmma_f32_16x16x32_bf16(false, A.v, false, B.v,
                                                    (short)0, acc, false, false);
    }
    __syncthreads();
  }

  // ---- reduce 4 k-groups per feature tile, apply ELU, store ----
  #pragma unroll
  for (int i = 0; i < 8; ++i) Red[wave][i][lane] = acc[i];
  __syncthreads();
  if (wave < 4) {
    #pragma unroll
    for (int i = 0; i < 8; ++i) {
      float v = Red[wave][i][lane] + Red[wave + 4][i][lane] +
                Red[wave + 8][i][lane] + Red[wave + 12][i][lane];
      v = (v > 0.f) ? v : (__expf(v) - 1.f);               // ELU (alpha=1)
      int rr = blockIdx.x * 16 + i + ((lane >> 4) << 3);   // C layout: vgpr i -> M=i / i+8
      int ff = wave * 16 + (lane & 15);
      out[(size_t)rr * NHID + ff] = v;
    }
  }
}

// ---------------- host-side launcher ------------------------------------------
extern "C" void kernel_launch(void* const* d_in, const int* in_sizes, int n_in,
                              void* d_out, int out_size, void* d_ws, size_t ws_size,
                              hipStream_t stream) {
  const float* x   = (const float*)d_in[0];        // (1, 8192, 128)
  const int*   adj = (const int*)d_in[1];          // (1, 8192, 8192)
  const float* W   = (const float*)d_in[2];        // (64, 128)
  const float* Wb  = (const float*)d_in[3];        // (64,)
  const float* aW  = (const float*)d_in[4];        // (1, 128)
  const float* aB  = (const float*)d_in[5];        // (1,)
  float* out = (float*)d_out;                      // (1, 8192, 64)

  float*    h       = (float*)d_ws;                // 8192*64 f32
  float*    s_src   = h + (size_t)NN * NHID;       // 8192
  float*    s_dst   = s_src + NN;                  // 8192
  unsigned* rowCnt  = (unsigned*)(s_dst + NN);     // 8192
  unsigned* rowBase = rowCnt + NN;                 // 8192
  __bf16*   hT      = (__bf16*)(rowBase + NN);     // 64*8192 bf16 (transposed, 1 MB)

  k_hidden  <<<NN / 4, 256, 0, stream>>>(x, W, Wb, h);
  k_scores  <<<NN / 256, 256, 0, stream>>>(h, aW, aB, s_src, s_dst, hT);
  k_rowcount<<<NN / 8, 256, 0, stream>>>(adj, rowCnt);
  k_scan    <<<1, 32, 0, stream>>>(rowCnt, rowBase);
  k_gat_main<<<NN / 16, 512, 0, stream>>>(adj, hT, s_src, s_dst, rowBase, out);
}
